// DeepSeekMoE_26843545600009
// MI455X (gfx1250) — matrix operand order repaired and checked by reading, not compiled, not run on hardware
//
#include <hip/hip_runtime.h>
#include <hip/hip_bf16.h>

#define N_TOKENS    8192
#define D_MODEL     2048
#define D_HIDDEN    2048
#define NUM_EXPERTS 16
#define TOP_K       2

#define MT 128   // tokens per block tile
#define NT 128   // hidden cols per block tile
#define KT 32    // K per WMMA step (bf16 16x16x32)
#define KSTEPS (D_MODEL / KT)
#define BPAD 8   // B row pad -> stride 40 ushorts = 80B (16B aligned, conflict-free)

typedef __attribute__((ext_vector_type(16))) __bf16 v16bf;
typedef __attribute__((ext_vector_type(2)))  __bf16 v2bf;
typedef __attribute__((ext_vector_type(8)))  float  v8f;

union FragU { uint4 q[2]; v16bf v; };

static __device__ inline unsigned pack2(float a, float b) {
    v2bf p; p[0] = (__bf16)a; p[1] = (__bf16)b;   // -> v_cvt_pk_bf16_f32
    return __builtin_bit_cast(unsigned, p);
}

// ---------------------------------------------------------------------------
// Kernel 0: zero output (atomic scatter-add target) and expert counters
// ---------------------------------------------------------------------------
__global__ void moe_zero_kernel(float* __restrict__ out, int* __restrict__ counts, int n4) {
    int i = blockIdx.x * blockDim.x + threadIdx.x;
    if (i < n4) ((float4*)out)[i] = make_float4(0.f, 0.f, 0.f, 0.f);
    if (blockIdx.x == 0 && threadIdx.x < NUM_EXPERTS) counts[threadIdx.x] = 0;
}

// ---------------------------------------------------------------------------
// Kernel 1: router — logits, top-2, softmax, build per-expert gather lists.
// One wave32 per token. Router weights (128KB) stay L2-resident.
// ---------------------------------------------------------------------------
__global__ __launch_bounds__(256) void moe_router_kernel(
    const float* __restrict__ x, const float* __restrict__ rw,
    int* __restrict__ counts, int* __restrict__ lists, float* __restrict__ gates)
{
    int wave = threadIdx.x >> 5;
    int lane = threadIdx.x & 31;
    int t = blockIdx.x * (blockDim.x >> 5) + wave;
    if (t >= N_TOKENS) return;

    float acc[NUM_EXPERTS];
#pragma unroll
    for (int e = 0; e < NUM_EXPERTS; ++e) acc[e] = 0.f;

    const float* xt = x + (size_t)t * D_MODEL;
    for (int k = lane; k < D_MODEL; k += 32) {
        float xv = xt[k];
#pragma unroll
        for (int e = 0; e < NUM_EXPERTS; ++e)
            acc[e] += xv * rw[e * D_MODEL + k];
    }
#pragma unroll
    for (int off = 16; off >= 1; off >>= 1) {
#pragma unroll
        for (int e = 0; e < NUM_EXPERTS; ++e)
            acc[e] += __shfl_xor(acc[e], off, 32);
    }
    if (lane == 0) {
        int b0 = 0;
#pragma unroll
        for (int e = 1; e < NUM_EXPERTS; ++e) if (acc[e] > acc[b0]) b0 = e;
        int b1 = (b0 == 0) ? 1 : 0;
#pragma unroll
        for (int e = 0; e < NUM_EXPERTS; ++e)
            if (e != b0 && acc[e] > acc[b1]) b1 = e;
        float ex = __expf(acc[b1] - acc[b0]);   // stable top-2 softmax
        float inv = 1.f / (1.f + ex);
        int p0 = atomicAdd(&counts[b0], 1);
        lists[b0 * N_TOKENS + p0] = t;
        gates[b0 * N_TOKENS + p0] = inv;
        int p1 = atomicAdd(&counts[b1], 1);
        lists[b1 * N_TOKENS + p1] = t;
        gates[b1 * N_TOKENS + p1] = ex * inv;
    }
}

// ---------------------------------------------------------------------------
// Kernel 2: per-expert gathered GEMM, bf16 WMMA, double-buffered LDS.
// Block = 256 threads (8 waves), tile = 128 tokens x 128 hidden.
// Wave grid 4x2: each wave owns 32 rows x 64 cols = 8 WMMAs per K-step.
// One barrier per K-step; global loads for step s+1 overlap WMMAs of step s.
// Gather/transpose addresses are precomputed as 32-bit element offsets so the
// hot loop uses saddr-base + vgpr-offset addressing with no 64-bit chains.
// ---------------------------------------------------------------------------
__global__ __launch_bounds__(256) void moe_expert_gemm_kernel(
    const float* __restrict__ x, const float* __restrict__ w,
    const int* __restrict__ counts, const int* __restrict__ lists,
    const float* __restrict__ gates, float* __restrict__ out)
{
    const int e = blockIdx.z;
    const int cnt = counts[e];
    const int m0 = blockIdx.x * MT;
    if (m0 >= cnt) return;
    const int n0 = blockIdx.y * NT;

    __shared__ __align__(16) unsigned short As[2][MT][KT];        // [buf][m][k]
    __shared__ __align__(16) unsigned short Bs[2][NT][KT + BPAD]; // [buf][n][k]
    __shared__ int sTok[MT];

    const int tid = threadIdx.x;
    if (tid < MT) {
        int gi = m0 + tid;
        if (gi >= cnt) gi = cnt - 1;               // clamp: rows >= cnt are don't-care
        sTok[tid] = lists[e * N_TOKENS + gi];
    }
    __syncthreads();

    const float* We = w + (size_t)e * D_MODEL * D_HIDDEN;

    const int lane  = tid & 31;
    const int wave  = tid >> 5;
    const int mW    = (wave >> 1) * 32;    // 2 row-subtiles per wave
    const int nW    = (wave & 1) * 64;     // 4 col-subtiles per wave
    const int l15   = lane & 15;
    const int khalf = lane >> 4;

    // loop-invariant 32-bit element offsets (x: 16M elems, We: 4M elems)
    int aoff[4], boff[4];
#pragma unroll
    for (int i = 0; i < 4; ++i) {
        int task = tid + i * 256;                  // 1024 staging tasks each
        int r  = task >> 3, c4 = (task & 7) * 4;   // A: row r, k-chunk c4
        aoff[i] = sTok[r] * D_MODEL + c4;
        int n  = task & (NT - 1);                  // B: col n, 4-row k-group kg
        int kg = (task >> 7) & 7;
        boff[i] = kg * 4 * D_HIDDEN + n0 + n;
    }

    auto loadA = [&](int k0, float4 (&av)[4]) {
#pragma unroll
        for (int i = 0; i < 4; ++i)
            av[i] = *(const float4*)(x + aoff[i] + k0);
    };
    auto storeA = [&](int buf, const float4 (&av)[4]) {
#pragma unroll
        for (int i = 0; i < 4; ++i) {
            int task = tid + i * 256;
            int r = task >> 3, c4 = (task & 7) * 4;
            *(uint2*)&As[buf][r][c4] =
                make_uint2(pack2(av[i].x, av[i].y), pack2(av[i].z, av[i].w));
        }
    };
    // B: 4(k) x 1(n) column chunks; transpose lives in the global access
    // pattern (lanes cover consecutive n -> coalesced b32 loads, strides in
    // the 24-bit instruction offset), LDS write is one contiguous b64.
    auto loadB = [&](int k0, float4 (&bv)[4]) {
#pragma unroll
        for (int i = 0; i < 4; ++i) {
            const float* p = We + boff[i] + k0 * D_HIDDEN;
            bv[i].x = p[0 * D_HIDDEN];
            bv[i].y = p[1 * D_HIDDEN];
            bv[i].z = p[2 * D_HIDDEN];
            bv[i].w = p[3 * D_HIDDEN];
        }
    };
    auto storeB = [&](int buf, const float4 (&bv)[4]) {
#pragma unroll
        for (int i = 0; i < 4; ++i) {
            int task = tid + i * 256;
            int n  = task & (NT - 1);
            int kg = (task >> 7) & 7;
            *(uint2*)&Bs[buf][n][kg * 4] =
                make_uint2(pack2(bv[i].x, bv[i].y), pack2(bv[i].z, bv[i].w));
        }
    };

    v8f acc[2][4] = {};

    // prologue: stage tile 0 into buffer 0
    {
        float4 av[4], bv[4];
        loadA(0, av);
        loadB(0, bv);
        storeA(0, av);
        storeB(0, bv);
    }
    __syncthreads();

    int buf = 0;
    for (int s = 0; s < KSTEPS; ++s) {
        float4 av[4], bv[4];
        const bool more = (s + 1 < KSTEPS);
        if (more) {                          // issue global loads early
            loadA((s + 1) * KT, av);
            loadB((s + 1) * KT, bv);
        }

        // A fragments (16-bit 16x32 layout): 2 x (2 x ds_load_b128)
        v16bf a[2];
#pragma unroll
        for (int mi = 0; mi < 2; ++mi) {
            const unsigned short* ap = &As[buf][mW + mi * 16 + l15][khalf * 16];
            FragU f;
            f.q[0] = *(const uint4*)ap;        // K kb..kb+7
            f.q[1] = *(const uint4*)(ap + 8);  // K kb+8..kb+15
            a[mi] = f.v;
        }
        // B fragments (32x16 layout, contiguous K in transposed LDS): 2 x b128 each
#pragma unroll
        for (int nb = 0; nb < 4; ++nb) {
            const unsigned short* bp = &Bs[buf][nW + nb * 16 + l15][khalf * 16];
            FragU f;
            f.q[0] = *(const uint4*)bp;        // K kb..kb+7
            f.q[1] = *(const uint4*)(bp + 8);  // K kb+8..kb+15
            v16bf b = f.v;
#pragma unroll
            for (int mi = 0; mi < 2; ++mi)
                acc[mi][nb] = __builtin_amdgcn_wmma_f32_16x16x32_bf16(
                    false, a[mi], false, b, (short)0, acc[mi][nb], false, false);
        }

        if (more) {                          // convert + fill the other buffer
            storeA(buf ^ 1, av);
            storeB(buf ^ 1, bv);
        }
        __syncthreads();
        buf ^= 1;
    }

    // epilogue: gate-scale + scatter-add (C/D layout: VGPR r, lanes>=16 -> M+8)
#pragma unroll
    for (int mi = 0; mi < 2; ++mi) {
#pragma unroll
        for (int r = 0; r < 8; ++r) {
            int row = mW + mi * 16 + r + (khalf << 3);
            int gi = m0 + row;
            if (gi < cnt) {
                int tok = sTok[row];
                float g = gates[e * N_TOKENS + gi];
                float* dst = out + (size_t)tok * D_HIDDEN + n0 + nW + l15;
#pragma unroll
                for (int nb = 0; nb < 4; ++nb)
                    atomicAdd(dst + nb * 16, g * acc[mi][nb][r]);
            }
        }
    }
}

// ---------------------------------------------------------------------------
extern "C" void kernel_launch(void* const* d_in, const int* in_sizes, int n_in,
                              void* d_out, int out_size, void* d_ws, size_t ws_size,
                              hipStream_t stream) {
    const float* x  = (const float*)d_in[0];
    const float* rw = (const float*)d_in[1];
    const float* we = (const float*)d_in[2];
    float* out = (float*)d_out;

    int*   counts = (int*)d_ws;                            // 16
    int*   lists  = counts + 16;                           // 16*8192
    float* gates  = (float*)(lists + NUM_EXPERTS * N_TOKENS);

    int n4 = out_size / 4;
    moe_zero_kernel<<<(n4 + 255) / 256, 256, 0, stream>>>(out, counts, n4);

    moe_router_kernel<<<N_TOKENS / 8, 256, 0, stream>>>(x, rw, counts, lists, gates);

    dim3 grid(N_TOKENS / MT, D_HIDDEN / NT, NUM_EXPERTS);  // 64 x 16 x 16
    moe_expert_gemm_kernel<<<grid, 256, 0, stream>>>(x, we, counts, lists, gates, out);
}
